// ViscoelasticMaterialModel_23270132809734
// MI455X (gfx1250) — matrix operand order repaired and checked by reading, not compiled, not run on hardware
//
#include <hip/hip_runtime.h>
#include <hip/hip_bf16.h>

#define M_DIM 501
#define H_DIM 128
#define KP2   1024   /* padded 2*M */
#define KP1   512    /* padded M   */
#define MPITCH (KP2 + 8)   /* LDS row pitch in bf16 elems (16B pad -> bank spread) */
#define NPITCH (KP1 + 8)
#define DT_C  0.01f

typedef __attribute__((ext_vector_type(16))) __bf16       v16bf;
typedef __attribute__((ext_vector_type(8)))  float        v8f;
typedef __attribute__((ext_vector_type(8)))  unsigned int v8u;

__device__ __forceinline__ unsigned short f2bf(float x) {
    union { float f; unsigned int u; } c; c.f = x;
    unsigned int u = c.u;
    return (unsigned short)((u + 0x7FFFu + ((u >> 16) & 1u)) >> 16);  // RNE
}

__device__ __forceinline__ float softplus(float x) {
    return (x > 20.f) ? x : log1pf(expf(x));
}

// ---------------------------------------------------------------------------
// Kernel 1: pack the three W1 matrices (fp32, KxH row-major) into bf16 frags
// for the v_wmma_f32_16x16x32_bf16 B operand, contiguous k order (matches A frags):
//   lane l: N = nt*16 + (l&15), K-half = (l>>4)*16, dword d holds K=base+2d,+2d+1
// One dword per thread; frag = 32 lanes x 8 dwords, contiguous -> coalesced
// 32B loads per lane in the GEMM kernel.
// Regions (tiles of 256 dwords): E w1: tiles 0..255, b w1: 256..511, n w1: 512..639
// ---------------------------------------------------------------------------
__global__ void pack_weights_kernel(const float* __restrict__ Ew1,
                                    const float* __restrict__ bw1,
                                    const float* __restrict__ nw1,
                                    unsigned int* __restrict__ packed) {
    int g = blockIdx.x * blockDim.x + threadIdx.x;
    if (g >= 640 * 256) return;
    int d    = g & 7;
    int lane = (g >> 3) & 31;
    int tile = g >> 8;

    const float* W; int K; int t = tile;
    if (t < 256)      { W = Ew1; K = 2 * M_DIM; }
    else if (t < 512) { W = bw1; K = 2 * M_DIM; t -= 256; }
    else              { W = nw1; K = M_DIM;     t -= 512; }
    int nt = t & 7, kc = t >> 3;
    int n  = nt * 16 + (lane & 15);
    int k0 = kc * 32 + ((lane >> 4) << 4) + 2 * d;
    int k1 = k0 + 1;
    unsigned short lo = (k0 < K) ? f2bf(W[(size_t)k0 * H_DIM + n]) : (unsigned short)0;
    unsigned short hi = (k1 < K) ? f2bf(W[(size_t)k1 * H_DIM + n]) : (unsigned short)0;
    packed[g] = (unsigned int)lo | ((unsigned int)hi << 16);
}

// A-fragment (16-bit A 16x32 layout): two ds_load_b128 per lane.
__device__ __forceinline__ v16bf load_afrag(const unsigned short* base, int pitch,
                                            int kc, int lane) {
    int off = (lane & 15) * pitch + kc * 32 + ((lane >> 4) ? 16 : 0);  // bf16 elems
    uint4 lo = *(const uint4*)(base + off);        // K +0..7   (or 16..23)
    uint4 hi = *(const uint4*)(base + off + 8);    // K +8..15  (or 24..31)
    v8u u = { lo.x, lo.y, lo.z, lo.w, hi.x, hi.y, hi.z, hi.w };
    return __builtin_bit_cast(v16bf, u);
}

// B-fragment: 32 contiguous bytes per lane from the packed region.
__device__ __forceinline__ v16bf load_bfrag(const unsigned int* region,
                                            int kc, int nt, int lane) {
    const unsigned int* p = region + ((((kc * 8 + nt) * 32) + lane) << 3);
    uint4 lo = *(const uint4*)p;
    uint4 hi = *(const uint4*)(p + 4);
    v8u u = { lo.x, lo.y, lo.z, lo.w, hi.x, hi.y, hi.z, hi.w };
    return __builtin_bit_cast(v16bf, u);
}

// ---------------------------------------------------------------------------
// Kernel 2: per 16-row tile of B, compute Emod / nuv / beta with WMMA.
// 256 threads = 8 waves; wave w owns H-columns [16w,16w+16) of ALL THREE nets.
// ---------------------------------------------------------------------------
__launch_bounds__(256)
__global__ void mlp_scalars_kernel(const float* __restrict__ Ein,
                                   const float* __restrict__ nuin,
                                   const unsigned int* __restrict__ packed,
                                   const float* __restrict__ Eb1, const float* __restrict__ Ew2, const float* __restrict__ Eb2,
                                   const float* __restrict__ nb1, const float* __restrict__ nw2, const float* __restrict__ nb2,
                                   const float* __restrict__ bb1, const float* __restrict__ bw2, const float* __restrict__ bb2,
                                   float* __restrict__ EmodOut,
                                   float* __restrict__ nuvOut,
                                   float* __restrict__ betaOut,
                                   int B) {
    __shared__ unsigned short mA[16 * MPITCH];   // [E|nu] rows, bf16, zero-padded to 1024
    __shared__ unsigned short nA[16 * NPITCH];   // nu rows, bf16, zero-padded to 512
    __shared__ float red[48];                    // [3][16] row reductions

    const int tid   = threadIdx.x;
    const int lane  = tid & 31;
    const int w     = tid >> 5;                  // wave id = N-tile id (0..7)
    const int tileB = blockIdx.x * 16;

    if (tid < 48) red[tid] = 0.f;

    // Stage A tiles into LDS (convert fp32 -> bf16, zero-pad).
    for (int idx = tid; idx < 16 * KP2; idx += 256) {
        int r = idx >> 10, k = idx & (KP2 - 1);
        int b = tileB + r;
        float v = 0.f;
        if (b < B) {
            if (k < M_DIM)            v = Ein[(size_t)b * M_DIM + k];
            else if (k < 2 * M_DIM)   v = nuin[(size_t)b * M_DIM + (k - M_DIM)];
        }
        mA[r * MPITCH + k] = f2bf(v);
    }
    for (int idx = tid; idx < 16 * KP1; idx += 256) {
        int r = idx >> 9, k = idx & (KP1 - 1);
        int b = tileB + r;
        float v = (b < B && k < M_DIM) ? nuin[(size_t)b * M_DIM + k] : 0.f;
        nA[r * NPITCH + k] = f2bf(v);
    }
    __syncthreads();

    const unsigned int* pE = packed;             // 256 tiles * 256 dwords
    const unsigned int* pb = packed + 65536;
    const unsigned int* pn = packed + 131072;

    v8f cE = {}; v8f cB = {}; v8f cN = {};
    for (int kc = 0; kc < KP2 / 32; ++kc) {
        v16bf a  = load_afrag(mA, MPITCH, kc, lane);
        v16bf bE = load_bfrag(pE, kc, w, lane);
        cE = __builtin_amdgcn_wmma_f32_16x16x32_bf16(false, a, false, bE, (short)0, cE, false, false);
        v16bf bB = load_bfrag(pb, kc, w, lane);
        cB = __builtin_amdgcn_wmma_f32_16x16x32_bf16(false, a, false, bB, (short)0, cB, false, false);
        if (kc < KP1 / 32) {
            v16bf an = load_afrag(nA, NPITCH, kc, lane);
            v16bf bN = load_bfrag(pn, kc, w, lane);
            cN = __builtin_amdgcn_wmma_f32_16x16x32_bf16(false, an, false, bN, (short)0, cN, false, false);
        }
    }

    // Epilogue: per lane, column n is fixed; rows = i + 8*(lane>=16).
    const int n = w * 16 + (lane & 15);
    const float eb1v = Eb1[n], ew2v = Ew2[n];
    const float bb1v = bb1[n], bw2v = bw2[n];
    const float nb1v = nb1[n], nw2v = nw2[n];
    const int rbase = (lane >> 4) ? 8 : 0;
#pragma unroll
    for (int i = 0; i < 8; ++i) {
        int r = i + rbase;
        atomicAdd(&red[r],      softplus(cE[i] + eb1v) * ew2v);
        atomicAdd(&red[16 + r], softplus(cB[i] + bb1v) * bw2v);
        atomicAdd(&red[32 + r], softplus(cN[i] + nb1v) * nw2v);
    }
    __syncthreads();

    if (tid < 16) {
        int b = tileB + tid;
        if (b < B) {
            float sE = red[tid]      + Eb2[0];
            float sB = red[16 + tid] + bb2[0];
            float sN = red[32 + tid] + nb2[0];
            EmodOut[b] = expf(sE);
            betaOut[b] = sB * sB * (1.f / 40000.f);
            nuvOut[b]  = expf(sN * sN * (1.f / 40000.f));
        }
    }
}

// ---------------------------------------------------------------------------
// Kernel 3: the time scan. One thread per batch row, float4 over T.
// Each lane streams a contiguous 2KB run of e / e_dot and writes stress / xi.
// ---------------------------------------------------------------------------
__global__ void scan_kernel(const float* __restrict__ e,
                            const float* __restrict__ ed,
                            const float* __restrict__ Emod,
                            const float* __restrict__ nuv,
                            const float* __restrict__ beta,
                            float* __restrict__ stress,
                            float* __restrict__ xiout,
                            int B, int T) {
    int b = blockIdx.x * blockDim.x + threadIdx.x;
    if (b >= B) return;
    const float Em = Emod[b], nv = nuv[b], btc = DT_C * beta[b];
    float xi = 0.f;
    const size_t base = (size_t)b * T;
    const float4* e4 = (const float4*)(e + base);
    const float4* d4 = (const float4*)(ed + base);
    float4* s4 = (float4*)(stress + base);
    float4* x4 = (float4*)(xiout + base);
    const int nT4 = T >> 2;
    for (int i = 0; i < nT4; ++i) {
        float4 ev = e4[i], dv = d4[i];
        float4 sv, xv; float q;
        q = ev.x - xi; sv.x = Em * ev.x + q + nv * dv.x; xv.x = xi; xi += btc * q;
        q = ev.y - xi; sv.y = Em * ev.y + q + nv * dv.y; xv.y = xi; xi += btc * q;
        q = ev.z - xi; sv.z = Em * ev.z + q + nv * dv.z; xv.z = xi; xi += btc * q;
        q = ev.w - xi; sv.w = Em * ev.w + q + nv * dv.w; xv.w = xi; xi += btc * q;
        s4[i] = sv; x4[i] = xv;
    }
    for (int t = nT4 << 2; t < T; ++t) {           // tail (T % 4 != 0)
        float ev = e[base + t], dv = ed[base + t];
        float q = ev - xi;
        stress[base + t] = Em * ev + q + nv * dv;
        xiout[base + t]  = xi;
        xi += btc * q;
    }
}

extern "C" void kernel_launch(void* const* d_in, const int* in_sizes, int n_in,
                              void* d_out, int out_size, void* d_ws, size_t ws_size,
                              hipStream_t stream) {
    const float* e   = (const float*)d_in[0];
    const float* ed  = (const float*)d_in[1];
    const float* E   = (const float*)d_in[2];
    const float* nu  = (const float*)d_in[3];
    const float* Ew1 = (const float*)d_in[4];
    const float* Eb1 = (const float*)d_in[5];
    const float* Ew2 = (const float*)d_in[6];
    const float* Eb2 = (const float*)d_in[7];
    const float* nw1 = (const float*)d_in[8];
    const float* nb1 = (const float*)d_in[9];
    const float* nw2 = (const float*)d_in[10];
    const float* nb2 = (const float*)d_in[11];
    const float* bw1 = (const float*)d_in[12];
    const float* bb1 = (const float*)d_in[13];
    const float* bw2 = (const float*)d_in[14];
    const float* bb2 = (const float*)d_in[15];

    const int B = in_sizes[2] / M_DIM;
    const int T = in_sizes[0] / B;

    // Workspace: [packed weights: 163840 dwords][Emod: B][nuv: B][beta: B]
    unsigned int* packed = (unsigned int*)d_ws;
    float* scal  = (float*)(packed + 163840);
    float* EmodA = scal;
    float* nuvA  = scal + B;
    float* betaA = scal + 2 * (size_t)B;

    pack_weights_kernel<<<640, 256, 0, stream>>>(Ew1, bw1, nw1, packed);

    mlp_scalars_kernel<<<(B + 15) / 16, 256, 0, stream>>>(
        E, nu, packed,
        Eb1, Ew2, Eb2, nb1, nw2, nb2, bb1, bw2, bb2,
        EmodA, nuvA, betaA, B);

    float* stress = (float*)d_out;
    float* xiout  = stress + (size_t)B * T;
    scan_kernel<<<(B + 255) / 256, 256, 0, stream>>>(
        e, ed, EmodA, nuvA, betaA, stress, xiout, B, T);
}
